// CLIPContrastiveCriterion_28535762714710
// MI455X (gfx1250) — matrix operand order repaired and checked by reading, not compiled, not run on hardware
//
#include <hip/hip_runtime.h>
#include <math.h>

// ---------------------------------------------------------------------------
// CLIP contrastive loss, N=16384, D=256, targets = arange(N).
// loss = 0.5*( mean_i[LSE_row_i - diag_i] + mean_j[LSE_col_j - diag_j] )
// Column-LSE of s*X@Y^T == row-LSE of s*Y@X^T, so one fused GEMM+online-LSE
// kernel is launched twice with swapped operands. GEMM runs on
// v_wmma_f32_16x16x32_bf16 using a bf16 hi/lo (bf16x2) split of the f32
// inputs: acc += Ahi*Bhi + Ahi*Blo + Alo*Bhi  (~16-bit mantissa fidelity).
// The two 16-col output tiles per iteration run on two independent
// accumulator chains, interleaved, so the matrix pipe always has an
// independent WMMA to issue.
// ---------------------------------------------------------------------------

#define NROWS 16384
#define DDIM  256

typedef __attribute__((ext_vector_type(16))) __bf16 bf16x16;
typedef __attribute__((ext_vector_type(8)))  float  f32x8;

union Frag {
    bf16x16 v;
    __bf16  e[16];
    uint4   q[2];
};

__device__ __forceinline__ f32x8 wmma_bf16(bf16x16 a, bf16x16 b, f32x8 c) {
    return __builtin_amdgcn_wmma_f32_16x16x32_bf16(
        /*neg_a=*/false, a, /*neg_b=*/false, b,
        /*c_mod=*/(short)0, c, /*reuse_a=*/false, /*reuse_b=*/false);
}

// ---------------------------------------------------------------------------
// Row-wise logsumexp of scale * (A @ B^T), A,B: [16384,256] f32 row-major.
// Grid: 128 blocks x 256 threads (8 waves). Wave w of block b owns rows
// [b*128 + w*16, +16). Loops over all 16384 columns in 32-col tiles.
// ---------------------------------------------------------------------------
__global__ __launch_bounds__(256) void lse_rows_kernel(
    const float* __restrict__ A, const float* __restrict__ B,
    const float* __restrict__ scale_p, float* __restrict__ lse_out)
{
    // B tile staged as bf16 hi/lo. Row stride 264 (=256+8) keeps the 16
    // lanes of a fragment load on distinct LDS banks (264*2B = 132 dwords,
    // bank step 4 per column).
    __shared__ __bf16 Bhi[32][264];
    __shared__ __bf16 Blo[32][264];

    const float scale = *scale_p;
    const int tid  = threadIdx.x;
    const int lane = tid & 31;
    const int wave = tid >> 5;
    const int half = lane >> 4;   // 0: rows 0-7 of tile, 1: rows 8-15
    const int mrow = lane & 15;   // A operand: M index; B/C operand: N index
    const int rowBase = blockIdx.x * 128 + wave * 16;

    // ---- Load this wave's 16x256 A block into WMMA A-fragments (hi/lo) ----
    // A operand layout (16-bit, 16x32): lane L holds M=L%16,
    // K = k0 + (L/16)*8 + 0..7 in elems 0-7 and K = k0+16+(L/16)*8+0..7 in 8-15.
    Frag ahi[8], alo[8];
    const float* arow = A + (size_t)(rowBase + mrow) * DDIM;
#pragma unroll
    for (int kc = 0; kc < 8; ++kc) {
        const int kb = kc * 32 + half * 8;
        float4 p0 = *(const float4*)(arow + kb);
        float4 p1 = *(const float4*)(arow + kb + 4);
        float4 p2 = *(const float4*)(arow + kb + 16);
        float4 p3 = *(const float4*)(arow + kb + 20);
        float ev[16] = {p0.x,p0.y,p0.z,p0.w, p1.x,p1.y,p1.z,p1.w,
                        p2.x,p2.y,p2.z,p2.w, p3.x,p3.y,p3.z,p3.w};
#pragma unroll
        for (int j = 0; j < 16; ++j) {
            float x  = ev[j];
            __bf16 h = (__bf16)x;                  // hi: RNE to bf16
            ahi[kc].e[j] = h;
            alo[kc].e[j] = (__bf16)(x - (float)h); // lo: residual
        }
    }

    // Online-LSE state, matching the C/D layout: m[v],s[v] on lanes 0-15
    // track row v, on lanes 16-31 row v+8, over columns N = lane%16 (mod 16).
    float m[8], s[8];
#pragma unroll
    for (int v = 0; v < 8; ++v) { m[v] = -INFINITY; s[v] = 0.0f; }

    for (int c0 = 0; c0 < NROWS; c0 += 32) {
        // ---- Cooperative stage of B[c0..c0+32) x K=256 into LDS hi/lo ----
        {
            const int col = tid >> 3;             // 32 cols, 8 threads/col
            const int kb  = (tid & 7) * 32;
            const float* brow = B + (size_t)(c0 + col) * DDIM + kb;
#pragma unroll
            for (int j = 0; j < 8; ++j) {
                float4 q = *(const float4*)(brow + j * 4);
                float qq[4] = {q.x, q.y, q.z, q.w};
#pragma unroll
                for (int e2 = 0; e2 < 4; ++e2) {
                    float x  = qq[e2];
                    __bf16 h = (__bf16)x;
                    Bhi[col][kb + j * 4 + e2] = h;
                    Blo[col][kb + j * 4 + e2] = (__bf16)(x - (float)h);
                }
            }
        }
        __syncthreads();

        // ---- Two 16x16 output tiles, K=256 in 8 chunks, 3 WMMAs/chunk,
        //      interleaved on two independent accumulator chains ----
        f32x8 acc0 = {0.f,0.f,0.f,0.f,0.f,0.f,0.f,0.f};
        f32x8 acc1 = {0.f,0.f,0.f,0.f,0.f,0.f,0.f,0.f};
        const int bcol0 = mrow;        // tile 0 column (N = lane%16)
        const int bcol1 = 16 + mrow;   // tile 1 column
#pragma unroll
        for (int kc = 0; kc < 8; ++kc) {
            // B operand layout (16-bit, 32x16): lane L holds K = kc*32+(L/16)*8+0..7
            // in elems 0-7 and K = kc*32+16+(L/16)*8+0..7 in 8-15, column L%16.
            const int kk = kc * 32 + half * 8;
            Frag bh0, bl0, bh1, bl1;
            bh0.q[0] = *(const uint4*)(&Bhi[bcol0][kk]);
            bh0.q[1] = *(const uint4*)(&Bhi[bcol0][kk + 16]);
            bh1.q[0] = *(const uint4*)(&Bhi[bcol1][kk]);
            bh1.q[1] = *(const uint4*)(&Bhi[bcol1][kk + 16]);
            bl0.q[0] = *(const uint4*)(&Blo[bcol0][kk]);
            bl0.q[1] = *(const uint4*)(&Blo[bcol0][kk + 16]);
            bl1.q[0] = *(const uint4*)(&Blo[bcol1][kk]);
            bl1.q[1] = *(const uint4*)(&Blo[bcol1][kk + 16]);
            acc0 = wmma_bf16(ahi[kc].v, bh0.v, acc0);
            acc1 = wmma_bf16(ahi[kc].v, bh1.v, acc1);
            acc0 = wmma_bf16(ahi[kc].v, bl0.v, acc0);
            acc1 = wmma_bf16(ahi[kc].v, bl1.v, acc1);
            acc0 = wmma_bf16(alo[kc].v, bh0.v, acc0);
            acc1 = wmma_bf16(alo[kc].v, bh1.v, acc1);
        }

        // ---- Online logsumexp update with x = scale * acc ----
#pragma unroll
        for (int v = 0; v < 8; ++v) {
            float x0 = scale * acc0[v];
            float nm = fmaxf(m[v], x0);
            s[v] = s[v] * __expf(m[v] - nm) + __expf(x0 - nm);
            m[v] = nm;
            float x1 = scale * acc1[v];
            nm = fmaxf(m[v], x1);
            s[v] = s[v] * __expf(m[v] - nm) + __expf(x1 - nm);
            m[v] = nm;
        }
        __syncthreads();
    }

    // ---- Combine the 16 lane-partials within each 16-lane half ----
#pragma unroll
    for (int off = 1; off < 16; off <<= 1) {
#pragma unroll
        for (int v = 0; v < 8; ++v) {
            float mo = __shfl_xor(m[v], off, 32);
            float so = __shfl_xor(s[v], off, 32);
            float nm = fmaxf(m[v], mo);
            s[v] = s[v] * __expf(m[v] - nm) + so * __expf(mo - nm);
            m[v] = nm;
        }
    }
    if (mrow == 0) {  // lanes 0 and 16 hold the fully-reduced rows
#pragma unroll
        for (int v = 0; v < 8; ++v)
            lse_out[rowBase + half * 8 + v] = m[v] + logf(s[v]);
    }
}

// ---------------------------------------------------------------------------
// diag[i] = scale * dot(X[i], Y[i])  (full f32 accuracy)
// ---------------------------------------------------------------------------
__global__ __launch_bounds__(256) void diag_kernel(
    const float* __restrict__ A, const float* __restrict__ B,
    const float* __restrict__ scale_p, float* __restrict__ dg)
{
    const int i = blockIdx.x * 256 + threadIdx.x;
    const float4* a4 = (const float4*)(A + (size_t)i * DDIM);
    const float4* b4 = (const float4*)(B + (size_t)i * DDIM);
    float acc = 0.0f;
#pragma unroll 8
    for (int j = 0; j < DDIM / 4; ++j) {
        float4 a = a4[j], b = b4[j];
        acc += a.x * b.x + a.y * b.y + a.z * b.z + a.w * b.w;
    }
    dg[i] = (*scale_p) * acc;
}

// ---------------------------------------------------------------------------
// loss = sum_i[(lse_r[i]-dg[i]) + (lse_c[i]-dg[i])] / (2N)
// ---------------------------------------------------------------------------
__global__ __launch_bounds__(256) void finish_kernel(
    const float* __restrict__ lr, const float* __restrict__ lc,
    const float* __restrict__ dg, float* __restrict__ out)
{
    __shared__ float red[256];
    float t = 0.0f;
    for (int i = threadIdx.x; i < NROWS; i += 256)
        t += (lr[i] - dg[i]) + (lc[i] - dg[i]);
    red[threadIdx.x] = t;
    __syncthreads();
#pragma unroll
    for (int off = 128; off > 0; off >>= 1) {
        if (threadIdx.x < off) red[threadIdx.x] += red[threadIdx.x + off];
        __syncthreads();
    }
    if (threadIdx.x == 0) out[0] = red[0] / (2.0f * (float)NROWS);
}

// ---------------------------------------------------------------------------
extern "C" void kernel_launch(void* const* d_in, const int* in_sizes, int n_in,
                              void* d_out, int out_size, void* d_ws, size_t ws_size,
                              hipStream_t stream) {
    const float* img     = (const float*)d_in[0];   // [16384,256] f32
    const float* txt     = (const float*)d_in[1];   // [16384,256] f32
    const float* scale_p = (const float*)d_in[2];   // [1] f32
    // d_in[3] = targets (arange) -- implied by the diagonal, unused.
    float* out = (float*)d_out;

    float* lse_r = (float*)d_ws;                    // 16384 f32
    float* lse_c = lse_r + NROWS;                   // 16384 f32
    float* dg    = lse_c + NROWS;                   // 16384 f32
    (void)in_sizes; (void)n_in; (void)out_size; (void)ws_size;

    // Row LSE of s*img@txt^T and (as rows of the swapped product) col LSE.
    lse_rows_kernel<<<NROWS / 128, 256, 0, stream>>>(img, txt, scale_p, lse_r);
    lse_rows_kernel<<<NROWS / 128, 256, 0, stream>>>(txt, img, scale_p, lse_c);
    diag_kernel<<<NROWS / 256, 256, 0, stream>>>(img, txt, scale_p, dg);
    finish_kernel<<<1, 256, 0, stream>>>(lse_r, lse_c, dg, out);
}
